// VMLA_Block_41566693491524
// MI455X (gfx1250) — matrix operand order repaired and checked by reading, not compiled, not run on hardware
//
#include <hip/hip_runtime.h>

// ---------------------------------------------------------------------------
// MLA transformer block for MI455X (gfx1250, wave32, WMMA f32_16x16x32_f16)
// ---------------------------------------------------------------------------

typedef __attribute__((ext_vector_type(16))) _Float16 v16h;
typedef __attribute__((ext_vector_type(8)))  float    v8f;

union HF { unsigned u[8]; v16h v; };

// problem sizes
constexpr int Bn   = 4;
constexpr int Sn   = 1024;
constexpr int Dn   = 1024;
constexpr int Hn   = 16;
constexpr int HDn  = 64;
constexpr int MVHn = 256;
constexpr int MLPn = 4096;
constexpr int Rn   = Bn * Sn;     // 4096 token rows

__device__ __forceinline__ int amapk(int j, int hlf) {
  // 16-bit A-matrix 16x32 lane/VGPR K mapping (ISA 7.12.2)
  return (j < 4 ? 2 * j : 16 + 2 * (j - 4)) + 8 * hlf;
}

__device__ __forceinline__ v8f vzero() {
  v8f z = {0.f, 0.f, 0.f, 0.f, 0.f, 0.f, 0.f, 0.f};
  return z;
}

__device__ __forceinline__ float gelu_exact(float x) {
  return 0.5f * x * (1.0f + erff(x * 0.70710678118654752f));
}

// ---------------------------------------------------------------------------
// f32 -> f16 conversion (weights)
// ---------------------------------------------------------------------------
__global__ __launch_bounds__(256) void k_cvt(const float* __restrict__ src,
                                             _Float16* __restrict__ dst, int n) {
  int i = blockIdx.x * 256 + threadIdx.x;
  if (i < n) dst[i] = (_Float16)src[i];
}

// ---------------------------------------------------------------------------
// LayerNorm (one row per block), writes f16
// ---------------------------------------------------------------------------
__global__ __launch_bounds__(256) void k_ln(const float* __restrict__ x,
                                            const float* __restrict__ w,
                                            _Float16* __restrict__ y) {
  int row = blockIdx.x;
  const float* xr = x + (size_t)row * Dn;
  float s = 0.f, s2 = 0.f;
  for (int i = threadIdx.x; i < Dn; i += 256) {
    float v = xr[i];
    s += v;
    s2 += v * v;
  }
#pragma unroll
  for (int o = 16; o >= 1; o >>= 1) {
    s  += __shfl_xor(s,  o, 32);
    s2 += __shfl_xor(s2, o, 32);
  }
  __shared__ float a1[8], a2[8];
  int wv = threadIdx.x >> 5;
  if ((threadIdx.x & 31) == 0) { a1[wv] = s; a2[wv] = s2; }
  __syncthreads();
  s = 0.f; s2 = 0.f;
#pragma unroll
  for (int k = 0; k < 8; ++k) { s += a1[k]; s2 += a2[k]; }
  float mu  = s * (1.0f / Dn);
  float var = s2 * (1.0f / Dn) - mu * mu;
  float rs  = rsqrtf(var + 1e-6f);
  for (int i = threadIdx.x; i < Dn; i += 256)
    y[(size_t)row * Dn + i] = (_Float16)((xr[i] - mu) * rs * w[i]);
}

// ---------------------------------------------------------------------------
// RoPE: src [Rn, 512] f32 (per-head 32 dims), writes f16 into dst[row, h*64+32+d]
// ---------------------------------------------------------------------------
__global__ __launch_bounds__(256) void k_rope(const float* __restrict__ src,
                                              const float* __restrict__ inv_freq,
                                              _Float16* __restrict__ dst) {
  int row = blockIdx.x;          // 0..Rn-1
  int p   = threadIdx.x;         // 0..255 : (h, d) pair
  int h   = p >> 4;
  int d   = p & 15;
  int pos = row & (Sn - 1);
  float ang = (float)pos * inv_freq[d];
  float c = cosf(ang), s = sinf(ang);
  const float* sr = src + (size_t)row * 512 + h * 32;
  float x1 = sr[d];
  float x2 = sr[d + 16];
  _Float16* dr = dst + (size_t)row * Dn + h * 64 + 32;
  dr[d]      = (_Float16)(x1 * c - x2 * s);
  dr[d + 16] = (_Float16)(x2 * c + x1 * s);
}

// ---------------------------------------------------------------------------
// Tiled WMMA GEMM:  C[M,N](+epi) = A[M,K] * W[N,K]^T
//   block = 256 threads = 8 waves ; block tile 64x128 ; wave tile 32x32
//   MODE bits: 1=f16 out, 2=GELU, 4=+bias[n], 8=resid[idx]+acc*scale[n], 16=split64 col remap
// ---------------------------------------------------------------------------
template <int MODE>
__global__ __launch_bounds__(256) void k_gemm(
    const _Float16* __restrict__ A, const _Float16* __restrict__ W,
    void* __restrict__ Cout, int M, int N, int K, int ldc,
    long batchA, long batchW, long batchC,
    const float* __restrict__ bias, const float* __restrict__ resid,
    const float* __restrict__ scale) {
  constexpr bool F16O = (MODE & 1)  != 0;
  constexpr bool GEL  = (MODE & 2)  != 0;
  constexpr bool BIA  = (MODE & 4)  != 0;
  constexpr bool RES  = (MODE & 8)  != 0;
  constexpr bool SPL  = (MODE & 16) != 0;
  (void)M;

  __shared__ _Float16 la[64][40];
  __shared__ _Float16 lb[128][40];

  int tid  = threadIdx.x;
  int wave = tid >> 5;
  int lane = tid & 31;
  int hlf  = lane >> 4;
  int l16  = lane & 15;
  int wr   = wave >> 2;       // 0..1
  int wc   = wave & 3;        // 0..3

  long zA = (long)blockIdx.z * batchA;
  long zW = (long)blockIdx.z * batchW;
  long zC = (long)blockIdx.z * batchC;
  int  m0 = blockIdx.y * 64;
  int  n0 = blockIdx.x * 128;

  v8f c[2][2];
  c[0][0] = vzero(); c[0][1] = vzero(); c[1][0] = vzero(); c[1][1] = vzero();

  int ar  = tid >> 2;          // 0..63
  int akg = (tid & 3) * 8;     // 0,8,16,24

  for (int k0 = 0; k0 < K; k0 += 32) {
    __syncthreads();
    *(uint4*)&la[ar][akg]      = *(const uint4*)(A + zA + (long)(m0 + ar) * K + k0 + akg);
    *(uint4*)&lb[ar][akg]      = *(const uint4*)(W + zW + (long)(n0 + ar) * K + k0 + akg);
    *(uint4*)&lb[ar + 64][akg] = *(const uint4*)(W + zW + (long)(n0 + ar + 64) * K + k0 + akg);
    // prime L2/L0 with the next K tile while this one computes (global_prefetch_b8)
    if (k0 + 32 < K) {
      __builtin_prefetch(A + zA + (long)(m0 + ar) * K + k0 + 32 + akg, 0, 1);
      __builtin_prefetch(W + zW + (long)(n0 + ar) * K + k0 + 32 + akg, 0, 1);
      __builtin_prefetch(W + zW + (long)(n0 + ar + 64) * K + k0 + 32 + akg, 0, 1);
    }
    __syncthreads();

    v16h af[2], bf[2];
#pragma unroll
    for (int a = 0; a < 2; ++a) {
      HF f;
      int lrow = wr * 32 + a * 16 + l16;
#pragma unroll
      for (int j = 0; j < 8; ++j)
        f.u[j] = *(const unsigned*)&la[lrow][amapk(j, hlf)];
      af[a] = f.v;
    }
#pragma unroll
    for (int bs = 0; bs < 2; ++bs) {
      HF f;
      int lcol = wc * 32 + bs * 16 + l16;
#pragma unroll
      for (int j = 0; j < 8; ++j)
        f.u[j] = *(const unsigned*)&lb[lcol][amapk(j, hlf)];
      bf[bs] = f.v;
    }
#pragma unroll
    for (int a = 0; a < 2; ++a)
#pragma unroll
      for (int bs = 0; bs < 2; ++bs)
        c[a][bs] = __builtin_amdgcn_wmma_f32_16x16x32_f16(
            false, af[a], false, bf[bs], (short)0, c[a][bs], false, false);
  }

  // epilogue
#pragma unroll
  for (int a = 0; a < 2; ++a)
#pragma unroll
    for (int bs = 0; bs < 2; ++bs)
#pragma unroll
      for (int r = 0; r < 8; ++r) {
        int gm = m0 + wr * 32 + a * 16 + r + 8 * hlf;
        int gn = n0 + wc * 32 + bs * 16 + l16;
        float v = c[a][bs][r];
        if constexpr (BIA) v += bias[gn];
        if constexpr (GEL) v = gelu_exact(v);
        long cn = SPL ? (long)((gn >> 5) * 64 + (gn & 31)) : (long)gn;
        long cidx = zC + (long)gm * ldc + cn;
        if constexpr (RES) v = resid[cidx] + v * scale[gn];
        if constexpr (F16O) ((_Float16*)Cout)[cidx] = (_Float16)v;
        else                ((float*)Cout)[cidx]    = v;
      }
}

// ---------------------------------------------------------------------------
// Flash-style attention with additive learned mask.
//   grid = (S/128, H, B), block = 256 (8 waves), each wave owns 16 q rows.
//   q/k/v are f16 [B*S, H*64]; mask f32 [B,S,S]; out f16 [B*S, H*64].
// ---------------------------------------------------------------------------
__global__ __launch_bounds__(256) void k_attn(
    const _Float16* __restrict__ qa, const _Float16* __restrict__ ka,
    const _Float16* __restrict__ va, const float* __restrict__ mask,
    _Float16* __restrict__ out) {
  __shared__ _Float16 lds_k[32][72];       // [key][d]
  __shared__ _Float16 lds_vt[64][40];      // [d][key]  (transposed)
  __shared__ _Float16 lds_p[8][16][40];    // per-wave P tile [m][key]

  int tid  = threadIdx.x;
  int wave = tid >> 5;
  int lane = tid & 31;
  int hlf  = lane >> 4;
  int l16  = lane & 15;
  int b = blockIdx.z, h = blockIdx.y;
  int qbase = blockIdx.x * 128 + wave * 16;
  long rowbase = (long)b * Sn + qbase;

  // load Q fragments (16 rows x 64 dims -> two 16x32 A-frags) from global
  v16h qf[2];
  {
    const _Float16* qrow = qa + (rowbase + l16) * Dn + h * 64;
#pragma unroll
    for (int dc = 0; dc < 2; ++dc) {
      HF f;
#pragma unroll
      for (int j = 0; j < 8; ++j)
        f.u[j] = *(const unsigned*)(qrow + amapk(j, hlf) + 32 * dc);
      qf[dc] = f.v;
    }
  }

  v8f acc[4];
  acc[0] = vzero(); acc[1] = vzero(); acc[2] = vzero(); acc[3] = vzero();
  float mrow[8], lrow[8];
#pragma unroll
  for (int r = 0; r < 8; ++r) { mrow[r] = -1e30f; lrow[r] = 0.f; }

  int sr = tid >> 3;            // 0..31 : key row being staged
  int sc = (tid & 7) * 8;       // 0,8,...,56 : d column group

  for (int kc = 0; kc < Sn; kc += 32) {
    __syncthreads();
    {
      const _Float16* ks = ka + ((long)b * Sn + kc + sr) * Dn + h * 64 + sc;
      *(uint4*)&lds_k[sr][sc] = *(const uint4*)ks;
      const _Float16* vs = va + ((long)b * Sn + kc + sr) * Dn + h * 64 + sc;
      uint4 vv = *(const uint4*)vs;
      const _Float16* ve = (const _Float16*)&vv;
#pragma unroll
      for (int e = 0; e < 8; ++e) lds_vt[sc + e][sr] = ve[e];
      // prefetch next key chunk (K and V rows) into cache
      if (kc + 32 < Sn) {
        __builtin_prefetch(ka + ((long)b * Sn + kc + 32 + sr) * Dn + h * 64 + sc, 0, 1);
        __builtin_prefetch(va + ((long)b * Sn + kc + 32 + sr) * Dn + h * 64 + sc, 0, 1);
      }
    }
    __syncthreads();

    // scores: two 16x16 key sub-tiles, contraction over d=64 (two WMMA each)
    v8f s[2];
#pragma unroll
    for (int kf = 0; kf < 2; ++kf) {
      v8f cc = vzero();
#pragma unroll
      for (int dc = 0; dc < 2; ++dc) {
        HF f;
#pragma unroll
        for (int j = 0; j < 8; ++j)
          f.u[j] = *(const unsigned*)&lds_k[kf * 16 + l16][amapk(j, hlf) + 32 * dc];
        cc = __builtin_amdgcn_wmma_f32_16x16x32_f16(
            false, qf[dc], false, f.v, (short)0, cc, false, false);
      }
      s[kf] = cc;
    }

    // scale + mask + online softmax, write P to wave-private LDS
#pragma unroll
    for (int r = 0; r < 8; ++r) {
      int qm = qbase + r + 8 * hlf;
      long mb = ((long)b * Sn + qm) * Sn + kc;
      float a0 = s[0][r] * 0.125f + mask[mb + l16];
      float a1 = s[1][r] * 0.125f + mask[mb + 16 + l16];
      float mx = fmaxf(a0, a1);
#pragma unroll
      for (int o = 8; o >= 1; o >>= 1) mx = fmaxf(mx, __shfl_xor(mx, o, 32));
      float mnew = fmaxf(mrow[r], mx);
      float resc = __expf(mrow[r] - mnew);
      mrow[r] = mnew;
      float e0 = __expf(a0 - mnew);
      float e1 = __expf(a1 - mnew);
      float rs = e0 + e1;
#pragma unroll
      for (int o = 8; o >= 1; o >>= 1) rs += __shfl_xor(rs, o, 32);
      lrow[r] = lrow[r] * resc + rs;
#pragma unroll
      for (int g = 0; g < 4; ++g) acc[g][r] *= resc;
      lds_p[wave][r + 8 * hlf][l16]      = (_Float16)e0;
      lds_p[wave][r + 8 * hlf][16 + l16] = (_Float16)e1;
    }

    // P (16x32) A-frag from LDS, then P @ V chunk (four 16-col groups)
    v16h pf;
    {
      HF f;
#pragma unroll
      for (int j = 0; j < 8; ++j)
        f.u[j] = *(const unsigned*)&lds_p[wave][l16][amapk(j, hlf)];
      pf = f.v;
    }
#pragma unroll
    for (int g = 0; g < 4; ++g) {
      HF f;
#pragma unroll
      for (int j = 0; j < 8; ++j)
        f.u[j] = *(const unsigned*)&lds_vt[g * 16 + l16][amapk(j, hlf)];
      acc[g] = __builtin_amdgcn_wmma_f32_16x16x32_f16(
          false, pf, false, f.v, (short)0, acc[g], false, false);
    }
  }

  // normalize + write
#pragma unroll
  for (int g = 0; g < 4; ++g)
#pragma unroll
    for (int r = 0; r < 8; ++r) {
      int qm = qbase + r + 8 * hlf;
      float v = acc[g][r] / lrow[r];
      out[((long)b * Sn + qm) * Dn + h * 64 + g * 16 + l16] = (_Float16)v;
    }
}

// ---------------------------------------------------------------------------
// Host orchestration
// ---------------------------------------------------------------------------
extern "C" void kernel_launch(void* const* d_in, const int* in_sizes, int n_in,
                              void* d_out, int out_size, void* d_ws, size_t ws_size,
                              hipStream_t stream) {
  (void)in_sizes; (void)n_in; (void)out_size; (void)ws_size;

  const float* input_q = (const float*)d_in[0];
  const float* ln_q_w  = (const float*)d_in[1];
  const float* invq    = (const float*)d_in[9];
  const float* invk    = (const float*)d_in[10];
  const float* mask_b1 = (const float*)d_in[12];
  const float* mask_b2 = (const float*)d_in[14];
  const float* ls_att  = (const float*)d_in[16];
  const float* ln2_w   = (const float*)d_in[17];
  const float* ls_mlp  = (const float*)d_in[20];

  char* ws = (char*)d_ws;
  size_t off = 0;
  auto alloc = [&](size_t bytes) -> void* {
    void* p = ws + off;
    off += (bytes + 255) & ~(size_t)255;
    return p;
  };

  // f16 weights
  _Float16* wencq  = (_Float16*)alloc((size_t)MVHn * Dn * 2);
  _Float16* wenckv = (_Float16*)alloc((size_t)MVHn * Dn * 2);
  _Float16* wq     = (_Float16*)alloc((size_t)512 * MVHn * 2);
  _Float16* wk     = (_Float16*)alloc((size_t)512 * MVHn * 2);
  _Float16* wv     = (_Float16*)alloc((size_t)Dn * MVHn * 2);
  _Float16* wqr    = (_Float16*)alloc((size_t)512 * MVHn * 2);
  _Float16* wkr    = (_Float16*)alloc((size_t)512 * Dn * 2);
  _Float16* wm1    = (_Float16*)alloc((size_t)2 * Sn * Sn * 2);
  _Float16* wm2    = (_Float16*)alloc((size_t)Sn * 2 * Sn * 2);
  _Float16* wo     = (_Float16*)alloc((size_t)Dn * Dn * 2);
  _Float16* wmlp1  = (_Float16*)alloc((size_t)MLPn * Dn * 2);
  _Float16* wmlp2  = (_Float16*)alloc((size_t)Dn * MLPn * 2);

  // activations (with reuse)
  _Float16* xq_h  = (_Float16*)alloc((size_t)Rn * Dn * 2);
  _Float16* zq_h  = (_Float16*)alloc((size_t)Rn * MVHn * 2);
  _Float16* zkv_h = (_Float16*)alloc((size_t)Rn * MVHn * 2);
  // qall..maskf form one contiguous 40MB region, reused later as hm (32MB)
  _Float16* qall  = (_Float16*)alloc((size_t)Rn * Dn * 2);      // 8MB
  _Float16* kall  = (_Float16*)alloc((size_t)Rn * Dn * 2);      // 8MB
  _Float16* vall  = (_Float16*)alloc((size_t)Rn * Dn * 2);      // 8MB
  float*    maskf = (float*)   alloc((size_t)Bn * Sn * Sn * 4); // 16MB
  float*    rtmp  = (float*)   alloc((size_t)Rn * 512 * 4);     // reused as xln (f16, same bytes)
  _Float16* smh   = (_Float16*)alloc((size_t)Bn * Sn * Sn * 2); // reused as x_att
  _Float16* h1h   = (_Float16*)alloc((size_t)Rn * 2 * Sn * 2);  // reused as x (f32, same bytes)

  _Float16* hm    = qall;                 // overlay on qall+kall+vall+maskf region
  _Float16* xatt  = smh;
  float*    xf    = (float*)h1h;
  _Float16* xln   = (_Float16*)rtmp;

  auto cvt = [&](const void* src, _Float16* dst, int n) {
    k_cvt<<<(n + 255) / 256, 256, 0, stream>>>((const float*)src, dst, n);
  };
  cvt(d_in[2],  wencq,  MVHn * Dn);       // first 256 rows of enc_q_w
  cvt(d_in[3],  wenckv, MVHn * Dn);
  cvt(d_in[4],  wq,     512 * MVHn);
  cvt(d_in[5],  wk,     512 * MVHn);
  cvt(d_in[6],  wv,     Dn * MVHn);
  cvt(d_in[7],  wqr,    512 * MVHn);
  cvt(d_in[8],  wkr,    512 * Dn);
  cvt(d_in[11], wm1,    2 * Sn * Sn);
  cvt(d_in[13], wm2,    Sn * 2 * Sn);
  cvt(d_in[15], wo,     Dn * Dn);
  cvt(d_in[18], wmlp1,  MLPn * Dn);
  cvt(d_in[19], wmlp2,  Dn * MLPn);

  // LN1
  k_ln<<<Rn, 256, 0, stream>>>(input_q, ln_q_w, xq_h);

  // latent compressions  zq, zkv : [4096,1024] x [256,1024]^T
  k_gemm<1><<<dim3(MVHn / 128, Rn / 64, 1), 256, 0, stream>>>(
      xq_h, wencq, zq_h, Rn, MVHn, Dn, MVHn, 0, 0, 0, nullptr, nullptr, nullptr);
  k_gemm<1><<<dim3(MVHn / 128, Rn / 64, 1), 256, 0, stream>>>(
      xq_h, wenckv, zkv_h, Rn, MVHn, Dn, MVHn, 0, 0, 0, nullptr, nullptr, nullptr);

  // content heads -> interleave into q_all/k_all [row, h*64 + 0..31]
  k_gemm<17><<<dim3(512 / 128, Rn / 64, 1), 256, 0, stream>>>(
      zq_h, wq, qall, Rn, 512, MVHn, Dn, 0, 0, 0, nullptr, nullptr, nullptr);
  k_gemm<17><<<dim3(512 / 128, Rn / 64, 1), 256, 0, stream>>>(
      zkv_h, wk, kall, Rn, 512, MVHn, Dn, 0, 0, 0, nullptr, nullptr, nullptr);

  // v heads [row, h*64+d] (v_proj output order already head-major)
  k_gemm<1><<<dim3(Dn / 128, Rn / 64, 1), 256, 0, stream>>>(
      zkv_h, wv, vall, Rn, Dn, MVHn, Dn, 0, 0, 0, nullptr, nullptr, nullptr);

  // RoPE q branch: qr = zq @ qr^T -> rtmp f32, rotate -> q_all[h*64+32..]
  k_gemm<0><<<dim3(512 / 128, Rn / 64, 1), 256, 0, stream>>>(
      zq_h, wqr, rtmp, Rn, 512, MVHn, 512, 0, 0, 0, nullptr, nullptr, nullptr);
  k_rope<<<Rn, 256, 0, stream>>>(rtmp, invq, qall);

  // RoPE k branch: kr = xq @ kr^T -> rtmp f32, rotate -> k_all[h*64+32..]
  k_gemm<0><<<dim3(512 / 128, Rn / 64, 1), 256, 0, stream>>>(
      xq_h, wkr, rtmp, Rn, 512, Dn, 512, 0, 0, 0, nullptr, nullptr, nullptr);
  k_rope<<<Rn, 256, 0, stream>>>(rtmp, invk, kall);

  // sm = qm @ km^T, batched over B  -> f16 [B,S,S]
  k_gemm<1><<<dim3(Sn / 128, Sn / 64, Bn), 256, 0, stream>>>(
      qall, kall, smh, Sn, Sn, Dn, Sn,
      (long)Sn * Dn, (long)Sn * Dn, (long)Sn * Sn, nullptr, nullptr, nullptr);

  // h1 = gelu(sm @ w1^T + b1) -> f16 [B*S, 2S]
  k_gemm<7><<<dim3(2 * Sn / 128, Rn / 64, 1), 256, 0, stream>>>(
      smh, wm1, h1h, Rn, 2 * Sn, Sn, 2 * Sn, 0, 0, 0, mask_b1, nullptr, nullptr);

  // mask = h1 @ w2^T + b2 -> f32 [B,S,S]
  k_gemm<4><<<dim3(Sn / 128, Rn / 64, 1), 256, 0, stream>>>(
      h1h, wm2, maskf, Rn, Sn, 2 * Sn, Sn, 0, 0, 0, mask_b2, nullptr, nullptr);

  // flash attention with additive mask -> x_att f16 [B*S, 1024]
  k_attn<<<dim3(Sn / 128, Hn, Bn), 256, 0, stream>>>(qall, kall, vall, maskf, xatt);

  // x = input + (x_att @ out_proj^T) * ls_att -> f32
  k_gemm<8><<<dim3(Dn / 128, Rn / 64, 1), 256, 0, stream>>>(
      xatt, wo, xf, Rn, Dn, Dn, Dn, 0, 0, 0, nullptr, input_q, ls_att);

  // LN2
  k_ln<<<Rn, 256, 0, stream>>>(xf, ln2_w, xln);

  // hm = gelu(x_ln @ mlp_w1^T) -> f16 [B*S, 4096]
  k_gemm<3><<<dim3(MLPn / 128, Rn / 64, 1), 256, 0, stream>>>(
      xln, wmlp1, hm, Rn, MLPn, Dn, MLPn, 0, 0, 0, nullptr, nullptr, nullptr);

  // out = x + ls_mlp * (hm @ mlp_w2^T) -> f32 d_out
  k_gemm<8><<<dim3(Dn / 128, Rn / 64, 1), 256, 0, stream>>>(
      hm, wmlp2, d_out, Rn, Dn, MLPn, Dn, 0, 0, 0, nullptr, xf, ls_mlp);
}
